// sLSTMLayer_35828617183868
// MI455X (gfx1250) — matrix operand order repaired and checked by reading, not compiled, not run on hardware
//
#include <hip/hip_runtime.h>

#define Bb 4
#define Ss 2048
#define Dd 768
#define NHh 4
#define DHh 192
#define Kc 4

typedef __attribute__((ext_vector_type(16))) __bf16 v16bf;
typedef __attribute__((ext_vector_type(8)))  float  v8f;

union BFTile { uint4 q[2]; v16bf v; };

typedef unsigned short u16;
typedef unsigned int   u32;
typedef unsigned long long u64;

// ---------- workspace layout (bytes) ----------
#define OFF_XB   0ull                                   // bf16 x        (B*S*D)
#define OFF_XCB  (OFF_XB  + (size_t)Bb*Ss*Dd*2)         // bf16 swish(conv(x))
#define OFF_WCAT (OFF_XCB + (size_t)Bb*Ss*Dd*2)         // bf16 [h][n=768][k=192] (N-major, transposed)
#define OFF_RCAT (OFF_WCAT + (size_t)NHh*768*192*2)     // bf16 [h][n=768][k=192]
#define OFF_G    (OFF_RCAT + (size_t)NHh*768*192*2)     // f32  [s][h][g][b][e]

#define RLDS_STRIDE 200          // halves per row (192 + pad), 400B: 16B aligned, bank-spread
#define GSTAGE_F32  (4*Bb*192)   // 3072 floats = 12288 B per (step, head)
#define SMEM_BYTES  ((768*RLDS_STRIDE + 16*RLDS_STRIDE)*2 + GSTAGE_F32*4)   // 325,888 B

__device__ __forceinline__ u16 f2bf(float f) {
  u32 u = __float_as_uint(f);
  u32 r = (u + 0x7FFFu + ((u >> 16) & 1u)) >> 16;   // round-to-nearest-even
  return (u16)r;
}

__device__ __forceinline__ void async_g2lds_b128(u32 lds_off, u64 gaddr) {
  asm volatile("global_load_async_to_lds_b128 %0, %1, off"
               :: "v"(lds_off), "v"(gaddr) : "memory");
}
__device__ __forceinline__ void wait_async0() {
  asm volatile("s_wait_asynccnt 0x0" ::: "memory");
}
__device__ __forceinline__ void wait_ds0() {
  asm volatile("s_wait_dscnt 0x0" ::: "memory");
}

// ---------------- kernel 1: weight prep (transpose to N-major bf16) ----------------
__global__ __launch_bounds__(256) void slstm_prep_w(
    const float* __restrict__ Wi, const float* __restrict__ Wf,
    const float* __restrict__ Wz, const float* __restrict__ Wo,
    const float* __restrict__ R,  u16* __restrict__ Wcat, u16* __restrict__ Rcat) {
  const int total = NHh * 768 * 192;
  int idx = blockIdx.x * blockDim.x + threadIdx.x;
  if (idx >= 2 * total) return;
  int j = (idx < total) ? idx : idx - total;
  int h  = j / (768 * 192);
  int rm = j % (768 * 192);
  int n  = rm / 192;
  int k  = rm % 192;
  int g  = n / 192;
  int e  = n % 192;
  if (idx < total) {
    const float* W = (g == 0) ? Wi : (g == 1) ? Wf : (g == 2) ? Wz : Wo;
    Wcat[j] = f2bf(W[(h * 192 + k) * 192 + e]);             // W[h][d=k][e]
  } else {
    Rcat[j] = f2bf(R[((g * NHh + h) * 192 + k) * 192 + e]); // R[g][h][d=k][e]
  }
}

// ---------------- kernel 2: causal depthwise conv + swish, bf16 casts ----------------
__global__ __launch_bounds__(256) void slstm_conv(
    const float* __restrict__ x, const float* __restrict__ ck,
    const float* __restrict__ cbias, u16* __restrict__ xb, u16* __restrict__ xcb) {
  int idx = blockIdx.x * blockDim.x + threadIdx.x;
  if (idx >= Bb * Ss * Dd) return;
  int d = idx % Dd;
  int s = (idx / Dd) % Ss;
  int b = idx / (Dd * Ss);
  float acc = cbias[d];
  #pragma unroll
  for (int j = 0; j < Kc; ++j) {
    int ss = s - (Kc - 1) + j;
    if (ss >= 0) acc += x[(b * Ss + ss) * Dd + d] * ck[j * Dd + d];
  }
  float sw = acc / (1.f + __expf(-acc));   // swish
  xcb[idx] = f2bf(sw);
  xb[idx]  = f2bf(x[idx]);
}

// ---------------- kernel 3: gate projections via WMMA ----------------
// grid = (512 M-tiles, 4 heads), block = 256 (8 waves). Wave w: n-tiles w*6..w*6+5.
// n-tiles 0..23 (gates i,f) read conv output; 24..47 (z,o) read x.
__global__ __launch_bounds__(256) void slstm_proj(
    const u16* __restrict__ xb, const u16* __restrict__ xcb,
    const u16* __restrict__ Wcat, const float* __restrict__ cell_bias,
    float* __restrict__ G) {
  const int h  = blockIdx.y;
  const int rt = blockIdx.x;
  const int lane  = threadIdx.x & 31;
  const int w     = threadIdx.x >> 5;
  const int lan15 = lane & 15;
  const int hi    = lane >> 4;
  const u16* A = (w < 4) ? xcb : xb;
  const int m = rt * 16 + lan15;          // A row for this lane

  v8f acc[6] = {};
  #pragma unroll
  for (int kk = 0; kk < 6; ++kk) {
    BFTile a;
    int col = h * 192 + kk * 32 + hi * 8;       // halves
    a.q[0] = *(const uint4*)&A[(size_t)m * Dd + col];
    a.q[1] = *(const uint4*)&A[(size_t)m * Dd + col + 16];
    #pragma unroll
    for (int j = 0; j < 6; ++j) {
      int nt = w * 6 + j;
      int n  = nt * 16 + lan15;                  // B col for this lane
      BFTile bt;
      const u16* Wp = &Wcat[((size_t)h * 768 + n) * 192 + kk * 32 + hi * 8];
      bt.q[0] = *(const uint4*)Wp;
      bt.q[1] = *(const uint4*)(Wp + 16);
      acc[j] = __builtin_amdgcn_wmma_f32_16x16x32_bf16(
          false, a.v, false, bt.v, (short)0, acc[j], false, false);
    }
  }
  // store: D layout — lane gives N, VGPR v gives M (+8 for hi lanes)
  #pragma unroll
  for (int j = 0; j < 6; ++j) {
    int nt = w * 6 + j;
    int n  = nt * 16 + lan15;
    int g  = n / 192, e = n % 192;
    float bias = cell_bias[g * Dd + h * 192 + e];
    #pragma unroll
    for (int v = 0; v < 8; ++v) {
      int r = rt * 16 + v + hi * 8;
      int b = r >> 11, s = r & (Ss - 1);
      G[(((size_t)(s * NHh + h) * 4 + g) * Bb + b) * 192 + e] = acc[j][v] + bias;
    }
  }
}

// ---------------- kernel 4: sequential sLSTM scan via WMMA + async G pipeline ----------------
// grid = 4 heads, block = 384 (12 waves). R_cat resident in LDS; gate block for step t+1
// DMA'd global->LDS (ASYNCcnt) while step t's WMMAs run. LDS use: 325,888 B of 320 KiB.
__global__ __launch_bounds__(384) void slstm_scan(
    const float* __restrict__ G, const u16* __restrict__ Rcat,
    float* __restrict__ out) {
  extern __shared__ u16 sm[];
  u16*   Rl = sm;                                            // 768 * RLDS_STRIDE halves
  u16*   hl = sm + 768 * RLDS_STRIDE;                        // 16  * RLDS_STRIDE halves
  float* gb = (float*)(sm + (768 + 16) * RLDS_STRIDE);       // 3072 f32 stage [g][b][e]
  const int h     = blockIdx.x;
  const int tid   = threadIdx.x;
  const int lane  = tid & 31;
  const int w     = tid >> 5;
  const int lan15 = lane & 15;
  const int hi    = lane >> 4;
  const int eo    = w * 16 + lan15;          // owned element index (lanes<16)

  // ---- one-shot staging: R_cat[h] -> LDS (async DMA, padded rows) ----
  const u16* Rg = Rcat + (size_t)h * 768 * 192;
  for (int i = tid; i < 768 * 24; i += 384) {          // 24 x 16B chunks per 192-half row
    int row = i / 24, c8 = i % 24;
    u32 loff = (u32)(size_t)&Rl[row * RLDS_STRIDE + c8 * 8];
    u64 gadr = (u64)(size_t)&Rg[row * 192 + c8 * 8];
    async_g2lds_b128(loff, gadr);
  }
  // ---- preload G(step 0) into stage (async DMA) ----
  {
    u64 g0 = (u64)(size_t)(G + (size_t)h * 4 * Bb * 192);
    u32 l0 = (u32)(size_t)gb + (u32)tid * 16u;
    async_g2lds_b128(l0, g0 + (u64)tid * 16u);
    async_g2lds_b128(l0 + 6144u, g0 + (u64)tid * 16u + 6144u);
  }
  // ---- h0 = 0 (incl. pad rows) ----
  for (int i = tid; i < 16 * RLDS_STRIDE / 2; i += 384)
    ((u32*)hl)[i] = 0u;
  wait_async0();
  __syncthreads();

  float cs[4] = {0.f, 0.f, 0.f, 0.f};
  float ns[4] = {0.f, 0.f, 0.f, 0.f};
  float ms[4] = {0.f, 0.f, 0.f, 0.f};

  for (int t = 0; t < Ss; ++t) {
    // C init = gate pre-activations for step t, from LDS stage
    v8f acc[4];
    #pragma unroll
    for (int g = 0; g < 4; ++g) {
      v8f c = {};
      if (lane < 16) {
        #pragma unroll
        for (int b = 0; b < 4; ++b) c[b] = gb[(g * Bb + b) * 192 + eo];
      }
      acc[g] = c;
    }
    wait_ds0();        // our stage reads complete before anyone overwrites the stage
    __syncthreads();

    // kick off DMA of G(t+1) into the stage; it lands under the WMMA work below
    if (t < Ss - 1) {
      u64 gn = (u64)(size_t)(G + (size_t)((t + 1) * NHh + h) * 4 * Bb * 192);
      u32 l0 = (u32)(size_t)gb + (u32)tid * 16u;
      async_g2lds_b128(l0, gn + (u64)tid * 16u);
      async_g2lds_b128(l0 + 6144u, gn + (u64)tid * 16u + 6144u);
    }
    if (t < Ss - 2) {  // warm L2 one step further ahead
      __builtin_prefetch(G + (size_t)((t + 2) * NHh + h) * 4 * Bb * 192 + tid, 0, 1);
    }

    // rec = h * R_cat  (M=16 padded batch dim, K=192, N: this wave's e-slice in all 4 gates)
    #pragma unroll
    for (int kk = 0; kk < 6; ++kk) {
      BFTile a;
      int ko = kk * 32 + hi * 8;
      a.q[0] = *(const uint4*)&hl[lan15 * RLDS_STRIDE + ko];
      a.q[1] = *(const uint4*)&hl[lan15 * RLDS_STRIDE + ko + 16];
      #pragma unroll
      for (int g = 0; g < 4; ++g) {
        int kb = kk * 32 + hi * 8;
        BFTile bt;
        const u16* Rp = &Rl[(g * 192 + w * 16 + lan15) * RLDS_STRIDE + kb];
        bt.q[0] = *(const uint4*)Rp;
        bt.q[1] = *(const uint4*)(Rp + 16);
        acc[g] = __builtin_amdgcn_wmma_f32_16x16x32_bf16(
            false, a.v, false, bt.v, (short)0, acc[g], false, false);
      }
    }
    __syncthreads();   // all waves done reading hl for step t

    if (lane < 16) {
      #pragma unroll
      for (int b = 0; b < 4; ++b) {
        float it = acc[0][b], ft = acc[1][b], zt = acc[2][b], ot = acc[3][b];
        float mn = fmaxf(ft + ms[b], it);
        float ia = __expf(it - mn);
        float fa = __expf(ft + ms[b] - mn);
        float cn = fa * cs[b] + ia * tanhf(zt);
        float nn = fa * ns[b] + ia;
        float hn = (cn / nn) / (1.f + __expf(-ot));
        cs[b] = cn; ns[b] = nn; ms[b] = mn;
        out[((size_t)(b * Ss + t)) * Dd + h * 192 + eo] = hn;   // (B,S,D) layout
        hl[b * RLDS_STRIDE + eo] = f2bf(hn);
      }
    }
    wait_async0();     // G(t+1) stage resident
    __syncthreads();   // hl + stage visible to all waves
  }
}

// ---------------- kernel 5: per-head group-norm, in place on d_out ----------------
__global__ __launch_bounds__(256) void slstm_gn(
    float* __restrict__ out, const float* __restrict__ gscale) {
  const int lane = threadIdx.x & 31;
  const int w    = threadIdx.x >> 5;
  int gid = blockIdx.x * 8 + w;               // (b*S+s)*NH + h, total 32768
  int hh  = gid & 3;
  int bs  = gid >> 2;
  float* p = out + (size_t)bs * Dd + hh * 192;
  float v[6];
  float s1 = 0.f, s2 = 0.f;
  #pragma unroll
  for (int j = 0; j < 6; ++j) {
    v[j] = p[lane + 32 * j];
    s1 += v[j];
    s2 += v[j] * v[j];
  }
  #pragma unroll
  for (int off = 16; off > 0; off >>= 1) {
    s1 += __shfl_xor(s1, off, 32);
    s2 += __shfl_xor(s2, off, 32);
  }
  float mu = s1 * (1.f / 192.f);
  float var = s2 * (1.f / 192.f) - mu * mu;
  float rs = rsqrtf(var + 1e-5f);
  #pragma unroll
  for (int j = 0; j < 6; ++j)
    p[lane + 32 * j] = (v[j] - mu) * rs * gscale[hh * 192 + lane + 32 * j];
}

extern "C" void kernel_launch(void* const* d_in, const int* in_sizes, int n_in,
                              void* d_out, int out_size, void* d_ws, size_t ws_size,
                              hipStream_t stream) {
  const float* x    = (const float*)d_in[0];
  const float* ck   = (const float*)d_in[1];
  const float* cb   = (const float*)d_in[2];
  const float* Wi   = (const float*)d_in[3];
  const float* Wf   = (const float*)d_in[4];
  const float* Wz   = (const float*)d_in[5];
  const float* Wo   = (const float*)d_in[6];
  const float* R    = (const float*)d_in[7];
  const float* cbia = (const float*)d_in[8];
  const float* gsc  = (const float*)d_in[9];
  float* out = (float*)d_out;
  char* ws = (char*)d_ws;

  u16*   xb   = (u16*)(ws + OFF_XB);
  u16*   xcb  = (u16*)(ws + OFF_XCB);
  u16*   Wcat = (u16*)(ws + OFF_WCAT);
  u16*   Rcat = (u16*)(ws + OFF_RCAT);
  float* G    = (float*)(ws + OFF_G);

  // 1) weight transpose/cast
  {
    int total = 2 * NHh * 768 * 192;
    slstm_prep_w<<<(total + 255) / 256, 256, 0, stream>>>(Wi, Wf, Wz, Wo, R, Wcat, Rcat);
  }
  // 2) conv + swish + bf16 casts
  slstm_conv<<<(Bb * Ss * Dd) / 256, 256, 0, stream>>>(x, ck, cb, xb, xcb);
  // 3) gate projections (WMMA)
  slstm_proj<<<dim3((Bb * Ss) / 16, NHh), 256, 0, stream>>>(xb, xcb, Wcat, cbia, G);
  // 4) recurrent scan (WMMA + async global->LDS pipeline, R resident in 320KB LDS)
  (void)hipFuncSetAttribute((const void*)slstm_scan,
                            hipFuncAttributeMaxDynamicSharedMemorySize, SMEM_BYTES);
  slstm_scan<<<NHh, 384, SMEM_BYTES, stream>>>(G, Rcat, out);
  // 5) group-norm in place
  slstm_gn<<<(Bb * Ss * NHh) / 8, 256, 0, stream>>>(out, gsc);
}
